// GPT_42580305772725
// MI455X (gfx1250) — matrix operand order repaired and checked by reading, not compiled, not run on hardware
//
#include <hip/hip_runtime.h>
#include <hip/hip_bf16.h>
#include <math.h>

// ---------------------------------------------------------------------------
// GPT-2ish forward for gfx1250 (MI455X). All GEMMs (qkv, proj, fc, fc2,
// lm_head) and both attention matmuls run on v_wmma_f32_16x16x32_bf16.
// GEMM is a 128x128 LDS-tiled, double-buffered kernel fed by
// global_load_async_to_lds_b128 (ASYNCcnt-tracked CDNA5 async copies).
// Attention is flash-style (no score materialization).
// ---------------------------------------------------------------------------

typedef __bf16 bf16;
typedef __attribute__((ext_vector_type(16))) __bf16 v16bf;
typedef __attribute__((ext_vector_type(8)))  __bf16 v8bf;
typedef __attribute__((ext_vector_type(8)))  float  v8f;

#define DMODEL 768
#define NH     12
#define HD     64
#define SEQ    2048
#define NB     2
#define NLAYER 2
#define VOCAB  32000
#define DFF    3072
#define MROWS  (NB * SEQ)   // 4096 token rows

// ---- helpers --------------------------------------------------------------

static __device__ __forceinline__ bf16 f2bf(float f) {
  unsigned u = __builtin_bit_cast(unsigned, f);
  unsigned r = u + 0x7FFFu + ((u >> 16) & 1u);   // round-to-nearest-even
  unsigned short s = (unsigned short)(r >> 16);
  return __builtin_bit_cast(bf16, s);
}

static __device__ __forceinline__ v16bf ld16(const bf16* p0, const bf16* p1) {
  v8bf a = *(const v8bf*)p0;
  v8bf b = *(const v8bf*)p1;
  return __builtin_shufflevector(a, b, 0, 1, 2, 3, 4, 5, 6, 7,
                                       8, 9, 10, 11, 12, 13, 14, 15);
}

static __device__ __forceinline__ v8f wmma_bf16(v16bf a, v16bf b, v8f c) {
  return __builtin_amdgcn_wmma_f32_16x16x32_bf16(false, a, false, b,
                                                 (short)0, c, false, false);
}

static __device__ __forceinline__ float gelu_exact(float v) {
  return 0.5f * v * (1.0f + erff(v * 0.70710678118654752f));
}

// CDNA5 async copy: LDS[lds_addr..+15] = MEM[g..g+15], tracked by ASYNCcnt.
static __device__ __forceinline__ void cp16(unsigned lds_addr, const bf16* g) {
  asm volatile("global_load_async_to_lds_b128 %0, %1, off"
               :: "v"(lds_addr), "v"((const void*)g) : "memory");
}

static __device__ __forceinline__ void wait_async_le4(bool more) {
  if (more) asm volatile("s_wait_asynccnt 0x4" ::: "memory");
  else      asm volatile("s_wait_asynccnt 0x0" ::: "memory");
}

// ---- elementwise kernels --------------------------------------------------

__global__ void k_cast(const float* __restrict__ s, bf16* __restrict__ d, int n) {
  int i = blockIdx.x * 256 + threadIdx.x;
  if (i < n) d[i] = f2bf(s[i]);
}

__global__ void k_embed(const int* __restrict__ idx, const float* __restrict__ wte,
                        const float* __restrict__ wpe, float* __restrict__ x) {
  int i  = blockIdx.x * 256 + threadIdx.x;       // over MROWS*DMODEL, exact grid
  int d  = i % DMODEL;
  int bt = i / DMODEL;
  int t  = bt % SEQ;
  x[i] = wte[(size_t)idx[bt] * DMODEL + d] + wpe[t * DMODEL + d];
}

// LayerNorm over D=768 (3 elems/thread, 256 threads/row), bf16 output.
__global__ __launch_bounds__(256) void k_ln(const float* __restrict__ x,
                                            const float* __restrict__ g,
                                            const float* __restrict__ bta,
                                            bf16* __restrict__ out) {
  __shared__ float red[256];
  const int tid = threadIdx.x;
  const float* xr = x + (size_t)blockIdx.x * DMODEL;
  float v0 = xr[tid], v1 = xr[tid + 256], v2 = xr[tid + 512];
  red[tid] = v0 + v1 + v2;
  __syncthreads();
  for (int st = 128; st > 0; st >>= 1) {
    if (tid < st) red[tid] += red[tid + st];
    __syncthreads();
  }
  const float mean = red[0] * (1.0f / DMODEL);
  __syncthreads();
  float d0 = v0 - mean, d1 = v1 - mean, d2 = v2 - mean;
  red[tid] = d0 * d0 + d1 * d1 + d2 * d2;
  __syncthreads();
  for (int st = 128; st > 0; st >>= 1) {
    if (tid < st) red[tid] += red[tid + st];
    __syncthreads();
  }
  const float rstd = rsqrtf(red[0] * (1.0f / DMODEL) + 1e-5f);
  bf16* orow = out + (size_t)blockIdx.x * DMODEL;
  orow[tid]       = f2bf(d0 * rstd * g[tid]       + bta[tid]);
  orow[tid + 256] = f2bf(d1 * rstd * g[tid + 256] + bta[tid + 256]);
  orow[tid + 512] = f2bf(d2 * rstd * g[tid + 512] + bta[tid + 512]);
}

// Split qkv f32 [bt, 3D] -> Q,K bf16 [B,H,T,64], Vt bf16 [B,H,64,T].
__global__ void k_split(const float* __restrict__ qkv, bf16* __restrict__ Q,
                        bf16* __restrict__ K, bf16* __restrict__ Vt) {
  int i  = blockIdx.x * 256 + threadIdx.x;       // over MROWS*DMODEL, exact grid
  int d  = i % DMODEL;
  int bt = i / DMODEL;
  int b  = bt / SEQ, t = bt - b * SEQ;
  int h  = d >> 6, dd = d & 63;
  const float* base = qkv + (size_t)bt * 3 * DMODEL;
  size_t bh = (size_t)(b * NH + h);
  Q[(bh * SEQ + t) * HD + dd] = f2bf(base[d]);
  K[(bh * SEQ + t) * HD + dd] = f2bf(base[DMODEL + d]);
  Vt[(bh * HD + dd) * SEQ + t] = f2bf(base[2 * DMODEL + d]);
}

// ---- WMMA GEMM: C[M,N] = A[M,K] * W[N,K]^T (+bias, epilogue) --------------
// Block tile 128Mx128N, K-step 32, double-buffered LDS filled by async copies.
// 8 waves; wave (mg = wid&3, ng = wid>>2) computes 32Mx64N.
// epi: 0 = Cf = v (f32)     1 = Cf += v (residual, f32)
//      2 = Cb = gelu(v)     3 = Cb = v (bf16)

__global__ __launch_bounds__(256) void k_gemm(const bf16* __restrict__ A,
                                              const bf16* __restrict__ W,
                                              const float* __restrict__ bias,
                                              float* __restrict__ Cf,
                                              bf16* __restrict__ Cb,
                                              int M, int N, int K, int epi) {
  __shared__ bf16 smA[2][128 * 32];   // [row][k], 8 KB per buffer
  __shared__ bf16 smB[2][128 * 32];   // [col][k], 8 KB per buffer

  const int tid  = threadIdx.x;
  const int lane = tid & 31;
  const int wid  = tid >> 5;
  const int mg = wid & 3;              // M group within block (x32)
  const int ng = wid >> 2;             // N group within block (x64)
  const int mA  = lane & 15;           // A frag: row within 16-tile
  const int kbA = (lane >> 4) * 8;     // A frag: K base {0,8}
  const int nB  = lane & 15;           // B frag: column within 16-tile
  const int kbB = (lane >> 4) * 8;     // B frag: K base {0,8}

  const bf16* Ag = A + (size_t)blockIdx.y * 128 * K;
  const bf16* Wg = W + (size_t)blockIdx.x * 128 * K;
  const unsigned baseA = (unsigned)(unsigned long long)(void*)&smA[0][0];
  const unsigned baseB = (unsigned)(unsigned long long)(void*)&smB[0][0];

  // async-stage one 128x32 A tile + 128x32 B tile (16 KB) into buffer `buf`
  auto stage = [&](int k0, int buf) {
    const unsigned dA = baseA + (unsigned)buf * 8192u;
    const unsigned dB = baseB + (unsigned)buf * 8192u;
    #pragma unroll
    for (int i = 0; i < 2; ++i) {
      const int seg = i * 256 + tid;          // 512 x 16B segments per tile
      const int row = seg >> 2;
      const int ks  = (seg & 3) * 8;          // element offset in K window
      cp16(dA + (unsigned)(row * 64 + ks * 2), Ag + (size_t)row * K + k0 + ks);
      cp16(dB + (unsigned)(row * 64 + ks * 2), Wg + (size_t)row * K + k0 + ks);
    }
  };

  v8f acc0[4] = {{}, {}, {}, {}};     // rows mg*32 + [0,16)
  v8f acc1[4] = {{}, {}, {}, {}};     // rows mg*32 + [16,32)

  const int nk = K >> 5;
  stage(0, 0);
  for (int it = 0; it < nk; ++it) {
    const int cur = it & 1;
    const bool more = (it + 1) < nk;
    if (more) stage((it + 1) << 5, cur ^ 1);
    wait_async_le4(more);       // retire current stage (async completes in order)
    __syncthreads();            // all waves' tiles visible

    const bf16* rA = &smA[cur][(mg * 32 + mA) * 32];
    const bf16* rB = &smB[cur][(ng * 64 + nB) * 32];
    v16bf a0 = ld16(rA + kbA,            rA + kbA + 16);
    v16bf a1 = ld16(rA + 16 * 32 + kbA,  rA + 16 * 32 + kbA + 16);
    #pragma unroll
    for (int j = 0; j < 4; ++j) {
      const bf16* rBj = rB + j * 16 * 32 + kbB;
      v16bf bfrag = ld16(rBj, rBj + 16);
      acc0[j] = wmma_bf16(a0, bfrag, acc0[j]);
      acc1[j] = wmma_bf16(a1, bfrag, acc1[j]);
    }
    __syncthreads();            // everyone done reading before buffer reuse
  }

  // ---- epilogue (C layout: rows mc..mc+7, col nc per 16x16 tile) ----------
  const int mc = (lane >> 4) * 8;
  const int nc = lane & 15;
  const int n0 = blockIdx.x * 128 + ng * 64;
  const float b0 = bias ? bias[n0 + nc]      : 0.0f;
  const float b1 = bias ? bias[n0 + 16 + nc] : 0.0f;
  const float b2 = bias ? bias[n0 + 32 + nc] : 0.0f;
  const float b3 = bias ? bias[n0 + 48 + nc] : 0.0f;

  #pragma unroll
  for (int mt = 0; mt < 2; ++mt) {
    const int m0 = blockIdx.y * 128 + mg * 32 + mt * 16;
    v8f* acc = mt ? acc1 : acc0;
    #pragma unroll
    for (int r = 0; r < 8; ++r) {
      const size_t row = (size_t)(m0 + mc + r) * N + n0 + nc;
      float v0 = acc[0][r] + b0, v1 = acc[1][r] + b1;
      float v2 = acc[2][r] + b2, v3 = acc[3][r] + b3;
      if (epi == 0) {
        Cf[row] = v0; Cf[row + 16] = v1; Cf[row + 32] = v2; Cf[row + 48] = v3;
      } else if (epi == 1) {
        Cf[row] += v0; Cf[row + 16] += v1; Cf[row + 32] += v2; Cf[row + 48] += v3;
      } else if (epi == 2) {
        Cb[row]      = f2bf(gelu_exact(v0)); Cb[row + 16] = f2bf(gelu_exact(v1));
        Cb[row + 32] = f2bf(gelu_exact(v2)); Cb[row + 48] = f2bf(gelu_exact(v3));
      } else {
        Cb[row]      = f2bf(v0); Cb[row + 16] = f2bf(v1);
        Cb[row + 32] = f2bf(v2); Cb[row + 48] = f2bf(v3);
      }
    }
  }
}

// ---- flash attention: one wave per 16-query tile --------------------------
// Q,K: [B,H,T,64] bf16   Vt: [B,H,64,T] bf16   y: [B,T,D] bf16
// 3072 wave-tasks, 8 waves/block -> 384 blocks.

__global__ __launch_bounds__(256) void k_attn(const bf16* __restrict__ Q,
                                              const bf16* __restrict__ K,
                                              const bf16* __restrict__ Vt,
                                              bf16* __restrict__ y) {
  __shared__ bf16 pls[8][16 * 32];       // per-wave P transpose tile (1 KB)
  const int lane = threadIdx.x & 31;
  const int wid  = threadIdx.x >> 5;
  const int task = blockIdx.x * 8 + wid; // [0, 3072)
  const int qt = task & 127;             // T/16 = 128 query tiles
  const int bh = task >> 7;              // [0, 24)
  const int b  = bh / NH;
  const int h  = bh - b * NH;
  const int q0 = qt * 16;

  const bf16* Qh = Q  + (size_t)bh * SEQ * HD;
  const bf16* Kh = K  + (size_t)bh * SEQ * HD;
  const bf16* Vh = Vt + (size_t)bh * HD * SEQ;

  const int mA  = lane & 15;
  const int kbA = (lane >> 4) * 8;
  const int nB  = lane & 15;
  const int kbB = (lane >> 4) * 8;
  const int mc  = (lane >> 4) * 8;
  const int nc  = lane & 15;

  // Q fragments for d-chunks [0,32) and [32,64)
  const bf16* Qrow = Qh + (size_t)(q0 + mA) * HD;
  v16bf qa0 = ld16(Qrow + kbA,      Qrow + kbA + 16);
  v16bf qa1 = ld16(Qrow + 32 + kbA, Qrow + 32 + kbA + 16);

  v8f o0 = {}, o1 = {}, o2 = {}, o3 = {};     // O tile: dims 0..63
  float rmax[8], rsum[8];
  #pragma unroll
  for (int r = 0; r < 8; ++r) { rmax[r] = -1e30f; rsum[r] = 0.0f; }

  bf16* P = &pls[wid][0];
  const int nchunk = (q0 + 47) >> 5;     // 32-key chunks covering 0..q0+15
  const float sc = 0.125f;               // 1/sqrt(64)

  for (int c = 0; c < nchunk; ++c) {
    const int j0 = c << 5;
    // S = Q K^T for keys [j0, j0+32)
    v8f s0 = {}, s1 = {};
    {
      const bf16* Kr0 = Kh + (size_t)(j0 + nB) * HD + kbB;
      const bf16* Kr1 = Kh + (size_t)(j0 + 16 + nB) * HD + kbB;
      s0 = wmma_bf16(qa0, ld16(Kr0,      Kr0 + 16), s0);
      s0 = wmma_bf16(qa1, ld16(Kr0 + 32, Kr0 + 48), s0);
      s1 = wmma_bf16(qa0, ld16(Kr1,      Kr1 + 16), s1);
      s1 = wmma_bf16(qa1, ld16(Kr1 + 32, Kr1 + 48), s1);
    }
    // online softmax per row (rows mc..mc+7 live in this 16-lane half)
    #pragma unroll
    for (int r = 0; r < 8; ++r) {
      const int q = q0 + mc + r;
      float v0 = (j0 + nc      <= q) ? s0[r] * sc : -1e30f;
      float v1 = (j0 + 16 + nc <= q) ? s1[r] * sc : -1e30f;
      float mx = fmaxf(v0, v1);
      mx = fmaxf(mx, __shfl_xor(mx, 1, 32));
      mx = fmaxf(mx, __shfl_xor(mx, 2, 32));
      mx = fmaxf(mx, __shfl_xor(mx, 4, 32));
      mx = fmaxf(mx, __shfl_xor(mx, 8, 32));
      const float newm = fmaxf(rmax[r], mx);
      const float p0 = __expf(v0 - newm);
      const float p1 = __expf(v1 - newm);
      float rs = p0 + p1;
      rs += __shfl_xor(rs, 1, 32);
      rs += __shfl_xor(rs, 2, 32);
      rs += __shfl_xor(rs, 4, 32);
      rs += __shfl_xor(rs, 8, 32);
      const float corr = __expf(rmax[r] - newm);
      rsum[r] = rsum[r] * corr + rs;
      rmax[r] = newm;
      o0[r] *= corr; o1[r] *= corr; o2[r] *= corr; o3[r] *= corr;
      P[(mc + r) * 32 + nc]      = f2bf(p0);
      P[(mc + r) * 32 + nc + 16] = f2bf(p1);
    }
    // C-layout -> A-layout transpose through per-wave LDS.
    // Same-wave LDS ops execute in order (ISA); fence stops compiler reorder.
    asm volatile("" ::: "memory");
    __builtin_amdgcn_wave_barrier();
    const bf16* Pr = P + mA * 32;
    v16bf pa = ld16(Pr + kbA, Pr + kbA + 16);
    asm volatile("" ::: "memory");
    __builtin_amdgcn_wave_barrier();
    // O += P * V   (Vt rows are dims -> contiguous B fragments)
    const bf16* Vr0 = Vh + (size_t)(nB)      * SEQ + j0 + kbB;
    const bf16* Vr1 = Vh + (size_t)(16 + nB) * SEQ + j0 + kbB;
    const bf16* Vr2 = Vh + (size_t)(32 + nB) * SEQ + j0 + kbB;
    const bf16* Vr3 = Vh + (size_t)(48 + nB) * SEQ + j0 + kbB;
    o0 = wmma_bf16(pa, ld16(Vr0, Vr0 + 16), o0);
    o1 = wmma_bf16(pa, ld16(Vr1, Vr1 + 16), o1);
    o2 = wmma_bf16(pa, ld16(Vr2, Vr2 + 16), o2);
    o3 = wmma_bf16(pa, ld16(Vr3, Vr3 + 16), o3);
  }

  #pragma unroll
  for (int r = 0; r < 8; ++r) {
    const float inv = 1.0f / rsum[r];
    const size_t off = (size_t)(b * SEQ + q0 + mc + r) * DMODEL + h * HD + nc;
    y[off]      = f2bf(o0[r] * inv);
    y[off + 16] = f2bf(o1[r] * inv);
    y[off + 32] = f2bf(o2[r] * inv);
    y[off + 48] = f2bf(o3[r] * inv);
  }
}

// ---------------------------------------------------------------------------

extern "C" void kernel_launch(void* const* d_in, const int* in_sizes, int n_in,
                              void* d_out, int out_size, void* d_ws, size_t ws_size,
                              hipStream_t stream) {
  (void)in_sizes; (void)n_in; (void)out_size; (void)ws_size;
  const int*   idx    = (const int*)  d_in[0];
  const float* wte    = (const float*)d_in[1];
  const float* wpe    = (const float*)d_in[2];
  const float* ln1_g  = (const float*)d_in[3];
  const float* ln1_b  = (const float*)d_in[4];
  const float* qkv_w  = (const float*)d_in[5];
  const float* qkv_b  = (const float*)d_in[6];
  const float* proj_w = (const float*)d_in[7];
  const float* proj_b = (const float*)d_in[8];
  const float* ln2_g  = (const float*)d_in[9];
  const float* ln2_b  = (const float*)d_in[10];
  const float* fc_w   = (const float*)d_in[11];
  const float* fc_b   = (const float*)d_in[12];
  const float* fc2_w  = (const float*)d_in[13];
  const float* fc2_b  = (const float*)d_in[14];
  const float* lnf_g  = (const float*)d_in[15];
  const float* lnf_b  = (const float*)d_in[16];
  float* logits = (float*)d_out;

  // workspace carve-out (~191 MB total), 256B aligned
  char* p = (char*)d_ws;
  auto take = [&](size_t bytes) -> void* {
    void* r = (void*)p;
    p += (bytes + 255) & ~(size_t)255;
    return r;
  };
  float* x    = (float*)take((size_t)MROWS * DMODEL * 4);            // 12.6 MB
  bf16*  hb   = (bf16*) take((size_t)MROWS * DMODEL * 2);            //  6.3 MB
  float* qkvo = (float*)take((size_t)MROWS * 3 * DMODEL * 4);        // 37.7 MB
  bf16*  Qb   = (bf16*) take((size_t)MROWS * DMODEL * 2);
  bf16*  Kb   = (bf16*) take((size_t)MROWS * DMODEL * 2);
  bf16*  Vtb  = (bf16*) take((size_t)MROWS * DMODEL * 2);            // 18.9 MB
  bf16*  yb   = (bf16*) take((size_t)MROWS * DMODEL * 2);            //  6.3 MB
  bf16*  ab   = (bf16*) take((size_t)MROWS * DFF * 2);               // 25.2 MB
  bf16*  xfb  = (bf16*) take((size_t)MROWS * DMODEL * 2);            //  6.3 MB
  bf16*  wteb = (bf16*) take((size_t)VOCAB * DMODEL * 2);            // 49.2 MB
  bf16*  qkvwb= (bf16*) take((size_t)NLAYER * 3 * DMODEL * DMODEL * 2);
  bf16*  projwb=(bf16*) take((size_t)NLAYER * DMODEL * DMODEL * 2);
  bf16*  fcwb = (bf16*) take((size_t)NLAYER * DFF * DMODEL * 2);
  bf16*  fc2wb= (bf16*) take((size_t)NLAYER * DMODEL * DFF * 2);     // 28.3 MB

  auto cast = [&](const float* s, bf16* d, size_t n) {
    k_cast<<<(unsigned)((n + 255) / 256), 256, 0, stream>>>(s, d, (int)n);
  };
  auto gemm = [&](const bf16* A, const bf16* W, const float* bias,
                  float* Cf, bf16* Cb, int M, int N, int K, int epi) {
    dim3 g(N / 128, M / 128);
    k_gemm<<<g, 256, 0, stream>>>(A, W, bias, Cf, Cb, M, N, K, epi);
  };

  // weights -> bf16 (weights all fit in 192 MB L2 for GEMM reuse)
  cast(wte,    wteb,  (size_t)VOCAB * DMODEL);
  cast(qkv_w,  qkvwb, (size_t)NLAYER * 3 * DMODEL * DMODEL);
  cast(proj_w, projwb,(size_t)NLAYER * DMODEL * DMODEL);
  cast(fc_w,   fcwb,  (size_t)NLAYER * DFF * DMODEL);
  cast(fc2_w,  fc2wb, (size_t)NLAYER * DMODEL * DFF);

  // embedding
  k_embed<<<(MROWS * DMODEL) / 256, 256, 0, stream>>>(idx, wte, wpe, x);

  for (int l = 0; l < NLAYER; ++l) {
    // h = ln1(x) in bf16
    k_ln<<<MROWS, 256, 0, stream>>>(x, ln1_g + l * DMODEL, ln1_b + l * DMODEL, hb);
    // qkv = h @ qkv_w^T + qkv_b  (f32)
    gemm(hb, qkvwb + (size_t)l * 3 * DMODEL * DMODEL, qkv_b + l * 3 * DMODEL,
         qkvo, nullptr, MROWS, 3 * DMODEL, DMODEL, 0);
    // split heads -> Q,K [B,H,T,64], Vt [B,H,64,T] (bf16)
    k_split<<<(MROWS * DMODEL) / 256, 256, 0, stream>>>(qkvo, Qb, Kb, Vtb);
    // flash attention -> y bf16 [B,T,D]
    k_attn<<<(NB * NH * (SEQ / 16)) / 8, 256, 0, stream>>>(Qb, Kb, Vtb, yb);
    // x += y @ proj_w^T + proj_b
    gemm(yb, projwb + (size_t)l * DMODEL * DMODEL, proj_b + l * DMODEL,
         x, nullptr, MROWS, DMODEL, DMODEL, 1);
    // h = ln2(x)
    k_ln<<<MROWS, 256, 0, stream>>>(x, ln2_g + l * DMODEL, ln2_b + l * DMODEL, hb);
    // a = gelu(h @ fc_w^T + fc_b) bf16
    gemm(hb, fcwb + (size_t)l * DFF * DMODEL, fc_b + l * DFF,
         nullptr, ab, MROWS, DFF, DMODEL, 2);
    // x += a @ fc2_w^T + fc2_b
    gemm(ab, fc2wb + (size_t)l * DMODEL * DFF, fc2_b + l * DMODEL,
         x, nullptr, MROWS, DMODEL, DFF, 1);
  }

  // final LN + tied lm_head: logits = lnf(x) @ wte^T  (f32 -> d_out, 524 MB)
  k_ln<<<MROWS, 256, 0, stream>>>(x, lnf_g, lnf_b, xfb);
  gemm(xfb, wteb, nullptr, logits, nullptr, MROWS, VOCAB, DMODEL, 0);
}
